// RNN_39393440039090
// MI455X (gfx1250) — matrix operand order repaired and checked by reading, not compiled, not run on hardware
//
#include <hip/hip_runtime.h>
#include <hip/hip_bf16.h>
#include <math.h>

// ---------------------------------------------------------------------------
// MI455X (gfx1250) stacked-LSTM.
//   x[16,32,64,256] -> TD-LSTM1(H=512) -> LSTM2(H=512) -> dense(1000)+softmax
// bf16 WMMA (v_wmma_f32_16x16x32_bf16), f32 accum/activations.
//   - weights packed once into WMMA B-operand lane layout (L2-resident, 3-4MB)
//   - x packed once into WMMA A-operand lane layout
//   - LSTM kernels: 1024-thread WGs (32 waves, 8 waves/SIMD). Each wave owns
//     4 gate n-tiles {32g+w} = all four gates of h-cols [16w,16w+16).
//   - c-state in VGPRs across all timesteps; h in LDS in A-operand swizzle.
//   - A laundered *integer offset* (not pointer!) stops LICM from hoisting
//     the t-invariant B-tile loads out of the timestep loop while preserving
//     the global address space (global_load_b128, not flat_load_b128).
// ---------------------------------------------------------------------------

typedef __attribute__((ext_vector_type(16))) __bf16 v16bf;
typedef __attribute__((ext_vector_type(8)))  float  v8f;

#define HD 512           // hidden
#define NG 2048          // 4*H gate width
#define B_SZ 16
#define TO 32
#define TI 64
#define FF 256
#define NCLS 1000

// round-to-nearest-even f32 -> bf16 bits
__device__ __forceinline__ unsigned short f2bf(float f) {
  unsigned int u = __float_as_uint(f);
  u += 0x7FFFu + ((u >> 16) & 1u);
  return (unsigned short)(u >> 16);
}
__device__ __forceinline__ float sigf(float x) {
  return 1.0f / (1.0f + __expf(-x));
}

// ---- WMMA operand lane maps (ISA 7.12.2, 16-bit data, wave32) -------------
// A (16x32, MxK): lane l, half e: M=l&15; v=e>>1, p=e&1;
//   K = (v<4 ? 2v+p : 16+2(v-4)+p) + (l>=16 ? 8 : 0)
// B (32x16, KxN): lane l, half e: N=l&15; K = (e&7) + ((e&8) ? 16 : 0) + (l>=16 ? 8 : 0)
// Inverse A map (used to scatter h into LDS):
//   lane = M + ((K&8)?16:0);  e = (K&7) + ((K&16)?8:0)
__device__ __forceinline__ int a_map_K(int lane, int e) {
  int v = e >> 1, p = e & 1;
  int k = (v < 4) ? (2 * v + p) : (16 + 2 * (v - 4) + p);
  return k + ((lane >= 16) ? 8 : 0);
}

// 4 B-operand tiles (wave's gate n-tiles {32g+w}) for one k-tile.
struct B4 { v16bf v[4]; };

__device__ __forceinline__ B4 load_b4(const unsigned short* pWz,
                                      int kt, int wave, int lane) {
  B4 r;
#pragma unroll
  for (int g = 0; g < 4; ++g)
    r.v[g] = *(const v16bf*)(
        pWz + ((size_t)kt * 128 + (32 * g + wave)) * 512 + lane * 16);
  return r;
}

__device__ __forceinline__ void wmma4(v8f acc[4], v16bf A, const B4& b) {
#pragma unroll
  for (int g = 0; g < 4; ++g)
    acc[g] = __builtin_amdgcn_wmma_f32_16x16x32_bf16(
        false, A, false, b.v[g], (short)0, acc[g], false, false);
}

// ---------------------------------------------------------------------------
// Kernel 1: pack [W;U] (f32, row-major [K,2048]) into bf16 B-operand tiles.
//   Wz1: K=768  (W1 256 rows + U1 512 rows) -> 24 ktiles x 128 ntiles
//   Wz2: K=1024 (W2 512 rows + U2 512 rows) -> 32 ktiles x 128 ntiles
// tile = 512 halves at [lane*16 + e].
// ---------------------------------------------------------------------------
__global__ __launch_bounds__(64) void pack_weights_kernel(
    const float* __restrict__ W1, const float* __restrict__ U1,
    const float* __restrict__ W2, const float* __restrict__ U2,
    unsigned short* __restrict__ pWz1, unsigned short* __restrict__ pWz2) {
  int blk = blockIdx.x;
  int tid = threadIdx.x;
  bool is2 = (blk >= 24 * 128);
  int t = is2 ? (blk - 24 * 128) : blk;
  int ktile = t >> 7;          // /128
  int ntile = t & 127;
  unsigned short* dst = (is2 ? pWz2 : pWz1) + (size_t)t * 512;
#pragma unroll
  for (int j = 0; j < 8; ++j) {
    int q = tid * 8 + j;
    int lane = q >> 4, e = q & 15;
    int N = 16 * ntile + (lane & 15);
    int K = 32 * ktile + (e & 7) + ((e & 8) ? 16 : 0) + ((lane >= 16) ? 8 : 0);
    float v;
    if (!is2) v = (K < FF) ? W1[(size_t)K * NG + N] : U1[(size_t)(K - FF) * NG + N];
    else      v = (K < HD) ? W2[(size_t)K * NG + N] : U2[(size_t)(K - HD) * NG + N];
    dst[q] = f2bf(v);
  }
}

// ---------------------------------------------------------------------------
// Kernel 2: pack x into bf16 A-operand tiles.
//   layout: pX[seqTile(32)][t(64)][ktile(8)][512 halves]
// ---------------------------------------------------------------------------
__global__ __launch_bounds__(64) void pack_x_kernel(
    const float* __restrict__ x, unsigned short* __restrict__ pX) {
  int blk = blockIdx.x;                  // 32*64*8 tiles
  int tid = threadIdx.x;
  int ktile = blk & 7;
  int t = (blk >> 3) & 63;
  int seqTile = blk >> 9;
  unsigned short* dst = pX + (size_t)blk * 512;
#pragma unroll
  for (int j = 0; j < 8; ++j) {
    int q = tid * 8 + j;
    int lane = q >> 4, e = q & 15;
    int M = lane & 15;
    int K = 32 * ktile + a_map_K(lane, e);
    int s = 16 * seqTile + M;            // sequence index b*32+to
    dst[q] = f2bf(x[((size_t)s * TI + t) * FF + K]);
  }
}

// ---------------------------------------------------------------------------
// Kernel 3: TimeDistributed LSTM1. 32 WGs x 1024 threads (32 waves).
// Each WG owns 16 sequences; per step: gates[16,2048] = [x_t|h] @ [W1;U1] + b
// Wave w owns gate n-tiles {32g+w} = gates i,f,g,o of h-cols [16w,16w+16).
// ---------------------------------------------------------------------------
__global__ __launch_bounds__(1024) void lstm1_kernel(
    const unsigned short* __restrict__ pX,
    const unsigned short* __restrict__ pWz,
    const float* __restrict__ b1,
    unsigned short* __restrict__ h1p) {
  __shared__ alignas(64) unsigned short hA[16 * 512];  // 16KB, A-swizzled h
  const int tid = threadIdx.x;
  const int wave = tid >> 5;   // 0..31
  const int lane = tid & 31;
  const int seqTile = blockIdx.x;

  for (int i = tid; i < 16 * 512; i += 1024) hA[i] = 0;  // h0 = 0
  __syncthreads();

  float bias[4];
#pragma unroll
  for (int g = 0; g < 4; ++g)
    bias[g] = b1[512 * g + 16 * wave + (lane & 15)];

  float cst[8];
#pragma unroll
  for (int r = 0; r < 8; ++r) cst[r] = 0.0f;

  const int rbase = (lane >= 16) ? 8 : 0;  // C/D: VGPR r holds row r(+8)
  size_t wzo = 0;                          // laundered offset (always 0)

  for (int t = 0; t < TI; ++t) {
    // Launder an integer offset: stops LICM from hoisting (and spilling) the
    // t-invariant B-tile loads, but keeps pWz's global address space intact.
    asm volatile("" : "+s"(wzo));
    const unsigned short* pWzt = pWz + wzo;

    v8f acc[4];
#pragma unroll
    for (int g = 0; g < 4; ++g)
#pragma unroll
      for (int r = 0; r < 8; ++r) acc[g][r] = bias[g];

    // prefetch next step's x tiles toward L2 (global_prefetch_b8)
    if (t + 1 < TI)
      __builtin_prefetch(pX + (((size_t)seqTile * TI + (t + 1)) * 8) * 512, 0, 1);

    const unsigned short* pXt =
        pX + (((size_t)seqTile * TI + t) * 8) * 512 + lane * 16;

    // ---- x part: kt 0..7, A from global (packed tiles).
#pragma unroll 2
    for (int kt = 0; kt < 8; ++kt) {
      v16bf A = *(const v16bf*)(pXt + (size_t)kt * 512);
      B4 b = load_b4(pWzt, kt, wave, lane);
      wmma4(acc, A, b);
    }
    // ---- h part: kt 8..23, A from LDS (ds_load_b128).
#pragma unroll 2
    for (int kt = 0; kt < 16; ++kt) {
      v16bf A = *(const v16bf*)&hA[kt * 512 + lane * 16];
      B4 b = load_b4(pWzt, 8 + kt, wave, lane);
      wmma4(acc, A, b);
    }
    __syncthreads();  // all hA reads complete before rewrite

#pragma unroll
    for (int r = 0; r < 8; ++r) {
      float iv = sigf(acc[0][r]);
      float fv = sigf(acc[1][r]);
      float gv = tanhf(acc[2][r]);
      float ov = sigf(acc[3][r]);
      float c = fv * cst[r] + iv * gv;
      cst[r] = c;
      float h = ov * tanhf(c);
      int row = r + rbase;                        // M
      int col = 16 * wave + (lane & 15);          // h column (K of next A)
      int kin = col & 31, ktl = col >> 5;
      int l2 = row + ((kin & 8) ? 16 : 0);        // inverse A map
      int e2 = (kin & 7) + ((kin & 16) ? 8 : 0);
      unsigned short hb = f2bf(h);
      hA[ktl * 512 + l2 * 16 + e2] = hb;
      if (t == TI - 1)
        h1p[(size_t)(16 * seqTile + row) * HD + col] = hb;  // plain bf16 [s,c]
    }
    __syncthreads();
  }
}

// ---------------------------------------------------------------------------
// Kernel 4: outer LSTM2. 1 WG x 1024 threads. K = 32 tiles (16 x + 16 h).
// x_t = h1[:, to, :] staged+swizzled into LDS each step.
// ---------------------------------------------------------------------------
__global__ __launch_bounds__(1024) void lstm2_kernel(
    const unsigned short* __restrict__ h1p,
    const unsigned short* __restrict__ pWz,
    const float* __restrict__ b2,
    float* __restrict__ h2) {
  __shared__ alignas(64) unsigned short xA[16 * 512];
  __shared__ alignas(64) unsigned short hA[16 * 512];
  const int tid = threadIdx.x;
  const int wave = tid >> 5;
  const int lane = tid & 31;

  for (int i = tid; i < 16 * 512; i += 1024) hA[i] = 0;

  float bias[4];
#pragma unroll
  for (int g = 0; g < 4; ++g)
    bias[g] = b2[512 * g + 16 * wave + (lane & 15)];

  float cst[8];
#pragma unroll
  for (int r = 0; r < 8; ++r) cst[r] = 0.0f;

  const int rbase = (lane >= 16) ? 8 : 0;
  const int srow = tid >> 6;        // staging row 0..15
  const int scol = tid & 63;        // staging col base
  size_t wzo = 0;                   // laundered offset (always 0)
  __syncthreads();

  for (int to = 0; to < TO; ++to) {
    asm volatile("" : "+s"(wzo));   // defeat LICM on B-tile loads
    const unsigned short* pWzt = pWz + wzo;

    // stage x_t = h1[b, to, :] into A-swizzled LDS (16 rows x 512 cols bf16)
#pragma unroll
    for (int i = 0; i < 8; ++i) {
      int c = scol + 64 * i;               // column 0..511
      unsigned short v = h1p[((size_t)srow * TO + to) * HD + c];
      int kin = c & 31, ktl = c >> 5;
      int l2 = srow + ((kin & 8) ? 16 : 0);
      int e2 = (kin & 7) + ((kin & 16) ? 8 : 0);
      xA[ktl * 512 + l2 * 16 + e2] = v;
    }
    __syncthreads();

    v8f acc[4];
#pragma unroll
    for (int g = 0; g < 4; ++g)
#pragma unroll
      for (int r = 0; r < 8; ++r) acc[g][r] = bias[g];

    // ---- x part: kt 0..15, A from xA (LDS).
#pragma unroll 2
    for (int kt = 0; kt < 16; ++kt) {
      v16bf A = *(const v16bf*)&xA[kt * 512 + lane * 16];
      B4 b = load_b4(pWzt, kt, wave, lane);
      wmma4(acc, A, b);
    }
    // ---- h part: kt 16..31, A from hA (LDS).
#pragma unroll 2
    for (int kt = 0; kt < 16; ++kt) {
      v16bf A = *(const v16bf*)&hA[kt * 512 + lane * 16];
      B4 b = load_b4(pWzt, 16 + kt, wave, lane);
      wmma4(acc, A, b);
    }
    __syncthreads();

#pragma unroll
    for (int r = 0; r < 8; ++r) {
      float iv = sigf(acc[0][r]);
      float fv = sigf(acc[1][r]);
      float gv = tanhf(acc[2][r]);
      float ov = sigf(acc[3][r]);
      float c = fv * cst[r] + iv * gv;
      cst[r] = c;
      float h = ov * tanhf(c);
      int row = r + rbase;
      int col = 16 * wave + (lane & 15);
      int kin = col & 31, ktl = col >> 5;
      int l2 = row + ((kin & 8) ? 16 : 0);
      int e2 = (kin & 7) + ((kin & 16) ? 8 : 0);
      hA[ktl * 512 + l2 * 16 + e2] = f2bf(h);
      if (to == TO - 1) h2[(size_t)row * HD + col] = h;
    }
    __syncthreads();
  }
}

// ---------------------------------------------------------------------------
// Kernel 5: dense + softmax. One WG per batch row (16 rows). 16.4 MFLOP total.
// ---------------------------------------------------------------------------
__global__ __launch_bounds__(256) void dense_softmax_kernel(
    const float* __restrict__ h2, const float* __restrict__ Wd,
    const float* __restrict__ bd, float* __restrict__ out) {
  __shared__ float hrow[HD];
  __shared__ float logits[NCLS];
  __shared__ float red[256];
  const int b = blockIdx.x;
  const int tid = threadIdx.x;
  hrow[tid] = h2[(size_t)b * HD + tid];
  hrow[tid + 256] = h2[(size_t)b * HD + tid + 256];
  __syncthreads();

  float lmax = -INFINITY;
#pragma unroll
  for (int j = 0; j < 4; ++j) {
    int c = tid + 256 * j;
    if (c < NCLS) {
      float d = bd[c];
      for (int k = 0; k < HD; ++k) d = fmaf(hrow[k], Wd[(size_t)k * NCLS + c], d);
      logits[c] = d;
      lmax = fmaxf(lmax, d);
    }
  }
  red[tid] = lmax;
  __syncthreads();
  for (int s = 128; s > 0; s >>= 1) {
    if (tid < s) red[tid] = fmaxf(red[tid], red[tid + s]);
    __syncthreads();
  }
  float m = red[0];
  __syncthreads();

  float lsum = 0.0f;
#pragma unroll
  for (int j = 0; j < 4; ++j) {
    int c = tid + 256 * j;
    if (c < NCLS) {
      float e = __expf(logits[c] - m);
      logits[c] = e;
      lsum += e;
    }
  }
  red[tid] = lsum;
  __syncthreads();
  for (int s = 128; s > 0; s >>= 1) {
    if (tid < s) red[tid] += red[tid + s];
    __syncthreads();
  }
  float inv = 1.0f / red[0];
#pragma unroll
  for (int j = 0; j < 4; ++j) {
    int c = tid + 256 * j;
    if (c < NCLS) out[(size_t)b * NCLS + c] = logits[c] * inv;
  }
}

// ---------------------------------------------------------------------------
// Workspace layout (bytes):
//   pX   : 32*64*8*512*2        = 16,777,216
//   pWz1 : 24*128*512*2         =  3,145,728
//   pWz2 : 32*128*512*2         =  4,194,304
//   h1p  : 512*512*2            =    524,288   (bf16)
//   h2   : 16*512*4             =     32,768   (f32)
//   total ~24.7 MB
// ---------------------------------------------------------------------------
extern "C" void kernel_launch(void* const* d_in, const int* in_sizes, int n_in,
                              void* d_out, int out_size, void* d_ws, size_t ws_size,
                              hipStream_t stream) {
  const float* x  = (const float*)d_in[0];
  const float* W1 = (const float*)d_in[1];
  const float* U1 = (const float*)d_in[2];
  const float* b1 = (const float*)d_in[3];
  const float* W2 = (const float*)d_in[4];
  const float* U2 = (const float*)d_in[5];
  const float* b2 = (const float*)d_in[6];
  const float* Wd = (const float*)d_in[7];
  const float* bd = (const float*)d_in[8];
  float* out = (float*)d_out;

  unsigned char* ws = (unsigned char*)d_ws;
  unsigned short* pX   = (unsigned short*)(ws);
  unsigned short* pWz1 = (unsigned short*)(ws + 16777216);
  unsigned short* pWz2 = (unsigned short*)(ws + 16777216 + 3145728);
  unsigned short* h1p  = (unsigned short*)(ws + 16777216 + 3145728 + 4194304);
  float*          h2   = (float*)(ws + 16777216 + 3145728 + 4194304 + 524288);

  pack_weights_kernel<<<dim3(24 * 128 + 32 * 128), dim3(64), 0, stream>>>(
      W1, U1, W2, U2, pWz1, pWz2);
  pack_x_kernel<<<dim3(32 * 64 * 8), dim3(64), 0, stream>>>(x, pX);
  lstm1_kernel<<<dim3(32), dim3(1024), 0, stream>>>(pX, pWz1, b1, h1p);
  lstm2_kernel<<<dim3(1), dim3(1024), 0, stream>>>(h1p, pWz2, b2, h2);
  dense_softmax_kernel<<<dim3(16), dim3(256), 0, stream>>>(h2, Wd, bd, out);
}
